// Block_70798240907438
// MI455X (gfx1250) — matrix operand order repaired and checked by reading, not compiled, not run on hardware
//
#include <hip/hip_runtime.h>

typedef __attribute__((ext_vector_type(16))) _Float16 v16h;
typedef __attribute__((ext_vector_type(8)))  _Float16 v8h;
typedef __attribute__((ext_vector_type(8)))  float    v8f;

#define N_PIX 14400
#define IMG_W 120

// ---------------------------------------------------------------------------
// channel max (adaptive_max_pool ks=1): per (b,c) max over 14400 pixels
// ---------------------------------------------------------------------------
__global__ __launch_bounds__(256) void channel_max_kernel(
    const float* __restrict__ src, float* __restrict__ out, int bstride) {
  const int b = blockIdx.x >> 6, c = blockIdx.x & 63;
  const float* p = src + (size_t)b * bstride + (size_t)c * N_PIX;
  float m = -3.4e38f;
  for (int i = threadIdx.x; i < N_PIX; i += 256) m = fmaxf(m, p[i]);
#pragma unroll
  for (int o = 16; o > 0; o >>= 1) m = fmaxf(m, __shfl_down(m, o, 32));
  __shared__ float s[8];
  const int lane = threadIdx.x & 31, w = threadIdx.x >> 5;
  if (lane == 0) s[w] = m;
  __syncthreads();
  if (threadIdx.x == 0) {
    float r = s[0];
#pragma unroll
    for (int i = 1; i < 8; ++i) r = fmaxf(r, s[i]);
    out[blockIdx.x] = r;
  }
}

// ---------------------------------------------------------------------------
// adaptive_max_pool ks=3 (regions of 40x40) -> gl9 (b,64,9)
// ---------------------------------------------------------------------------
__global__ __launch_bounds__(256) void pool3_kernel(
    const float* __restrict__ src, float* __restrict__ out, int bstride) {
  const int b = blockIdx.x >> 6, c = blockIdx.x & 63;
  const float* p = src + (size_t)b * bstride + (size_t)c * N_PIX;
  float m[9];
#pragma unroll
  for (int r = 0; r < 9; ++r) m[r] = -3.4e38f;
  for (int i = threadIdx.x; i < N_PIX; i += 256) {
    int y = i / IMG_W, x = i - y * IMG_W;
    int r = (y / 40) * 3 + (x / 40);
    m[r] = fmaxf(m[r], p[i]);
  }
#pragma unroll
  for (int r = 0; r < 9; ++r)
#pragma unroll
    for (int o = 16; o > 0; o >>= 1) m[r] = fmaxf(m[r], __shfl_down(m[r], o, 32));
  __shared__ float s[8][9];
  const int lane = threadIdx.x & 31, w = threadIdx.x >> 5;
  if (lane == 0) {
#pragma unroll
    for (int r = 0; r < 9; ++r) s[w][r] = m[r];
  }
  __syncthreads();
  if (threadIdx.x < 9) {
    float acc = s[0][threadIdx.x];
#pragma unroll
    for (int ww = 1; ww < 8; ++ww) acc = fmaxf(acc, s[ww][threadIdx.x]);
    out[(size_t)blockIdx.x * 9 + threadIdx.x] = acc;
  }
}

// ---------------------------------------------------------------------------
// 1x1 content gates: w16[b,oc,64] = sigmoid(gate_c[c]+gate_o[oc]) * wconv
// ---------------------------------------------------------------------------
__global__ __launch_bounds__(256) void gates1x1_kernel(
    const float* __restrict__ gmax, const float* __restrict__ ce,
    const float* __restrict__ gd, const float* __restrict__ gd2,
    const float* __restrict__ ci, const float* __restrict__ wconv,
    _Float16* __restrict__ w16, int OC, int CI_OUT) {
  const int b = blockIdx.x, tid = threadIdx.x;
  __shared__ float g[64], gc[64], go[192];
  if (tid < 64) {
    float v = fmaxf(gmax[b * 64 + tid] * ce[0], 0.f);
    g[tid] = v;
    gc[tid] = v * gd[0];
  }
  __syncthreads();
  if (tid < OC) {
    const int grp = tid / CI_OUT, o = tid - grp * CI_OUT;
    float s = 0.f;
#pragma unroll
    for (int gi = 0; gi < 8; ++gi) s += g[grp * 8 + gi] * ci[o * 8 + gi];
    go[tid] = fmaxf(s, 0.f) * gd2[0];
  }
  __syncthreads();
  for (int idx = tid; idx < OC * 64; idx += 256) {
    const int oc = idx >> 6, c = idx & 63;
    const float dyn = 1.f / (1.f + expf(-(gc[c] + go[oc])));
    w16[(size_t)b * OC * 64 + idx] = (_Float16)(dyn * wconv[idx]);
  }
}

// ---------------------------------------------------------------------------
// 3x3 content gates for fc2: w16[b,64,576] (k = c*9 + 3*di + dj)
// ---------------------------------------------------------------------------
__global__ __launch_bounds__(256) void gates3x3_kernel(
    const float* __restrict__ gl9, const float* __restrict__ ce,
    const float* __restrict__ gd, const float* __restrict__ gd2,
    const float* __restrict__ ci, const float* __restrict__ wconv,
    _Float16* __restrict__ w16) {
  const int b = blockIdx.x, tid = threadIdx.x;
  __shared__ float gl[576], r1[320], gc[576], r2[320], go[576];
  for (int i = tid; i < 576; i += 256) gl[i] = gl9[(size_t)b * 576 + i];
  __syncthreads();
  for (int i = tid; i < 320; i += 256) {       // r1[c*5+l] = relu(ce2)
    const int c = i / 5, l = i - 5 * c;
    float s = 0.f;
#pragma unroll
    for (int j = 0; j < 9; ++j) s += gl[c * 9 + j] * ce[l * 9 + j];
    r1[i] = fmaxf(s, 0.f);
  }
  __syncthreads();
  for (int i = tid; i < 576; i += 256) {       // gate_c[c*9+j]
    const int c = i / 9, j = i - 9 * c;
    float s = 0.f;
#pragma unroll
    for (int l = 0; l < 5; ++l) s += r1[c * 5 + l] * gd[j * 5 + l];
    gc[i] = s;
  }
  for (int i = tid; i < 320; i += 256) {       // r2[oc*5+l] = relu(oc_t)
    const int oc = i / 5, l = i - 5 * oc;
    const int grp = oc >> 3, o = oc & 7;
    float s = 0.f;
#pragma unroll
    for (int g = 0; g < 8; ++g) s += r1[(grp * 8 + g) * 5 + l] * ci[o * 8 + g];
    r2[i] = fmaxf(s, 0.f);
  }
  __syncthreads();
  for (int i = tid; i < 576; i += 256) {       // gate_o[oc*9+j]
    const int oc = i / 9, j = i - 9 * oc;
    float s = 0.f;
#pragma unroll
    for (int l = 0; l < 5; ++l) s += r2[oc * 5 + l] * gd2[j * 5 + l];
    go[i] = s;
  }
  __syncthreads();
  for (int idx = tid; idx < 64 * 576; idx += 256) {
    const int oc = idx / 576, rem = idx - oc * 576;
    const int c = rem / 9, j = rem - 9 * c;
    const float dyn = 1.f / (1.f + expf(-(gc[c * 9 + j] + go[oc * 9 + j])));
    w16[(size_t)b * 64 * 576 + idx] = (_Float16)(dyn * wconv[idx]);
  }
}

// ---------------------------------------------------------------------------
// WMMA GEMM: dst(b, OC, 14400) = W16(b, OC, KTOT) @ im2col?(src)(b, KTOT, 14400)
// Block: 256 threads (8 waves), tile M=16 x N=256, K staged in LDS chunks of 32.
// LDS layouts are pre-swizzled so every fragment is two 16B ds_load_b128:
//   sWs: per K-chunk, per row m (40-half stride), local k stored at [k]
//        (contiguous: lane half hi holds k = hi*16 + e, same as sXc)
//   sXc: column-major, 40-half column stride; column col holds k = hi*16+e
// EPI==1 fuses exact GELU.
// ---------------------------------------------------------------------------
template <int OC, int KTOT, bool IM2COL, int EPI>
__global__ __launch_bounds__(256) void gemm_wmma_kernel(
    const _Float16* __restrict__ wmat, const float* __restrict__ src,
    float* __restrict__ dst, int src_bstride, int dst_bstride) {
  constexpr int NT = 256, KC = 32;
  constexpr int XSTR = 40;                 // halves per column (80B, 16B aligned)
  constexpr int WSTR = 40;                 // halves per (chunk,m) group
  constexpr int WCHUNK = 16 * WSTR;        // 640 halves per K-chunk
  __shared__ _Float16 sWs[(KTOT / KC) * WCHUNK];
  __shared__ _Float16 sXc[NT * XSTR];
  const int b = blockIdx.z;
  const int m0 = blockIdx.y * 16;
  const int n0 = blockIdx.x * NT;
  const int tid = threadIdx.x;
  const int lane = tid & 31, wv = tid >> 5;
  const int mrow = lane & 15, hi = lane >> 4;

  // --- stage W, pre-permuted into fragment order -------------------------
  const _Float16* wb = wmat + (size_t)b * OC * KTOT;
  for (int i = tid; i < 16 * KTOT; i += 256) {
    const int m = i / KTOT, k = i - m * KTOT;
    const int kc = k >> 5, kl = k & 31;
    const int e = (kl & 7) | ((kl >> 4) << 3);
    const int h = (kl >> 3) & 1;
    sWs[kc * WCHUNK + m * WSTR + kl] = wb[(size_t)(m0 + m) * KTOT + k];
  }
  const float* sb = src + (size_t)b * src_bstride;
  v8f acc0 = {}, acc1 = {};

  for (int kbase = 0; kbase < KTOT; kbase += KC) {
    const int kc = kbase >> 5;
    // prefetch next K chunk (global_prefetch_b8)
    if (!IM2COL && kbase + KC < KTOT) {
      const int pcol = n0 + ((tid & 7) << 5);
      if (pcol < N_PIX)
        __builtin_prefetch(&sb[(size_t)(kbase + KC + (tid >> 3)) * N_PIX + pcol], 0, 1);
    }
    // --- stage X chunk, branchless (clamp address, cndmask-select zero) --
#pragma unroll 4
    for (int i = tid; i < KC * NT; i += 256) {
      const int kk = i >> 8, col = i & 255;
      const int gcol = n0 + col;
      const bool cin = gcol < N_PIX;
      const int gc = cin ? gcol : (N_PIX - 1);
      float v;
      bool keep;
      if (!IM2COL) {
        v = sb[(size_t)(kbase + kk) * N_PIX + gc];
        keep = cin;
      } else {
        const int k = kbase + kk;
        const int c = k / 9, r = k - 9 * c;
        const int di = r / 3 - 1, dj = r - (r / 3) * 3 - 1;
        const int y = gc / IMG_W, x = gc - y * IMG_W;
        const int yy = y + di, xx = x + dj;
        keep = cin & (yy >= 0) & (yy < IMG_W) & (xx >= 0) & (xx < IMG_W);
        const int yc = min(max(yy, 0), IMG_W - 1);
        const int xc = min(max(xx, 0), IMG_W - 1);
        v = sb[(size_t)c * N_PIX + yc * IMG_W + xc];
      }
      v = keep ? v : 0.f;
      sXc[col * XSTR + kk] = (_Float16)v;
    }
    __syncthreads();
    // --- fragments: two ds_load_b128 each --------------------------------
    const _Float16* wp = &sWs[kc * WCHUNK + mrow * WSTR + hi * 16];
    const v8h w_lo = *(const v8h*)wp;
    const v8h w_hi = *(const v8h*)(wp + 8);
    const v16h af = __builtin_shufflevector(w_lo, w_hi, 0, 1, 2, 3, 4, 5, 6, 7,
                                            8, 9, 10, 11, 12, 13, 14, 15);
    const int colb = wv * 32;
    const _Float16* xp0 = &sXc[(colb + mrow) * XSTR + hi * 16];
    const v8h b0l = *(const v8h*)xp0;
    const v8h b0h = *(const v8h*)(xp0 + 8);
    const v16h bf0 = __builtin_shufflevector(b0l, b0h, 0, 1, 2, 3, 4, 5, 6, 7,
                                             8, 9, 10, 11, 12, 13, 14, 15);
    const _Float16* xp1 = &sXc[(colb + 16 + mrow) * XSTR + hi * 16];
    const v8h b1l = *(const v8h*)xp1;
    const v8h b1h = *(const v8h*)(xp1 + 8);
    const v16h bf1 = __builtin_shufflevector(b1l, b1h, 0, 1, 2, 3, 4, 5, 6, 7,
                                             8, 9, 10, 11, 12, 13, 14, 15);
    acc0 = __builtin_amdgcn_wmma_f32_16x16x32_f16(false, af, false, bf0, (short)0,
                                                  acc0, false, false);
    acc1 = __builtin_amdgcn_wmma_f32_16x16x32_f16(false, af, false, bf1, (short)0,
                                                  acc1, false, false);
    __syncthreads();
  }
  // --- epilogue + store (C/D layout: VGPR r -> M = r + hi*8, N = lane%16) --
  float* db = dst + (size_t)b * dst_bstride;
  const int colb = wv * 32;
#pragma unroll
  for (int r = 0; r < 8; ++r) {
    const int row = m0 + r + hi * 8;
    const int c0 = n0 + colb + mrow;
    float v0 = acc0[r], v1 = acc1[r];
    if (EPI == 1) {
      v0 = v0 * 0.5f * (1.f + erff(v0 * 0.70710678118654752f));
      v1 = v1 * 0.5f * (1.f + erff(v1 * 0.70710678118654752f));
    }
    if (c0 < N_PIX) db[(size_t)row * N_PIX + c0] = v0;
    if (c0 + 16 < N_PIX) db[(size_t)row * N_PIX + c0 + 16] = v1;
  }
}

// ---------------------------------------------------------------------------
// k1/v1 = (add ? k1 : 0) + k_new/v_new   (k,v slices of qkv)
// ---------------------------------------------------------------------------
__global__ __launch_bounds__(256) void kv_accum_kernel(
    const float* __restrict__ qkv, float* __restrict__ k1, float* __restrict__ v1,
    int add) {
  const size_t idx = (size_t)blockIdx.x * 256 + threadIdx.x;
  const size_t per_b = (size_t)64 * N_PIX;
  if (idx >= (size_t)8 * per_b) return;
  const size_t b = idx / per_b, r = idx - b * per_b;
  const float* base = qkv + b * (size_t)192 * N_PIX;
  const float kn = base[(size_t)64 * N_PIX + r];
  const float vn = base[(size_t)128 * N_PIX + r];
  if (add) { k1[idx] += kn; v1[idx] += vn; }
  else     { k1[idx] = kn;  v1[idx] = vn; }
}

// ---------------------------------------------------------------------------
// attention scores: one block per (b, head). Fused q.k dots + q/k row norms,
// scale by head_w, softmax over d. attnW layout (b,8,8,8).
// ---------------------------------------------------------------------------
__global__ __launch_bounds__(256) void attn_scores_kernel(
    const float* __restrict__ qkv, const float* __restrict__ k1,
    const float* __restrict__ hw, float* __restrict__ attnW) {
  const int b = blockIdx.x >> 3, h = blockIdx.x & 7;
  const float* qb = qkv + (size_t)b * 192 * N_PIX + (size_t)h * 8 * N_PIX;
  const float* kb = k1 + (size_t)b * 64 * N_PIX + (size_t)h * 8 * N_PIX;
  float acc[80];
#pragma unroll
  for (int i = 0; i < 80; ++i) acc[i] = 0.f;
  for (int p = threadIdx.x; p < N_PIX; p += 256) {
    float qv[8], kv[8];
#pragma unroll
    for (int c = 0; c < 8; ++c) qv[c] = qb[(size_t)c * N_PIX + p];
#pragma unroll
    for (int d = 0; d < 8; ++d) kv[d] = kb[(size_t)d * N_PIX + p];
#pragma unroll
    for (int c = 0; c < 8; ++c) {
      acc[64 + c] = fmaf(qv[c], qv[c], acc[64 + c]);
      acc[72 + c] = fmaf(kv[c], kv[c], acc[72 + c]);
#pragma unroll
      for (int d = 0; d < 8; ++d) acc[c * 8 + d] = fmaf(qv[c], kv[d], acc[c * 8 + d]);
    }
  }
#pragma unroll
  for (int i = 0; i < 80; ++i) {
#pragma unroll
    for (int o = 16; o > 0; o >>= 1) acc[i] += __shfl_down(acc[i], o, 32);
  }
  __shared__ float part[8][80];
  __shared__ float tot[64], nrm[16], S[64];
  const int lane = threadIdx.x & 31, w = threadIdx.x >> 5;
  if (lane == 0) {
#pragma unroll
    for (int i = 0; i < 80; ++i) part[w][i] = acc[i];
  }
  __syncthreads();
  if (threadIdx.x < 80) {
    float s = 0.f;
#pragma unroll
    for (int ww = 0; ww < 8; ++ww) s += part[ww][threadIdx.x];
    if (threadIdx.x < 64) tot[threadIdx.x] = s;
    else nrm[threadIdx.x - 64] = fmaxf(sqrtf(s), 1e-12f);
  }
  __syncthreads();
  if (threadIdx.x < 64) {
    const int c = threadIdx.x >> 3, d = threadIdx.x & 7;
    S[threadIdx.x] = tot[threadIdx.x] / (nrm[c] * nrm[8 + d]) * hw[(h * 8 + c) * 8 + d];
  }
  __syncthreads();
  if (threadIdx.x < 8) {
    const int c = threadIdx.x;
    float mx = -3.4e38f;
#pragma unroll
    for (int d = 0; d < 8; ++d) mx = fmaxf(mx, S[c * 8 + d]);
    float e[8], sum = 0.f;
#pragma unroll
    for (int d = 0; d < 8; ++d) { e[d] = expf(S[c * 8 + d] - mx); sum += e[d]; }
    const float inv = 1.f / sum;
#pragma unroll
    for (int d = 0; d < 8; ++d)
      attnW[(size_t)(b * 8 + h) * 64 + c * 8 + d] = e[d] * inv;
  }
}

// ---------------------------------------------------------------------------
// fused: out = attn @ v, then cur += LayerNorm_c(out) (per pixel)
// ---------------------------------------------------------------------------
__global__ __launch_bounds__(256) void attn_apply_ln_kernel(
    const float* __restrict__ attnW, const float* __restrict__ v1,
    const float* __restrict__ lnw, const float* __restrict__ lnb,
    float* __restrict__ cur) {
  const int b = blockIdx.y;
  const int pix = blockIdx.x * 256 + threadIdx.x;
  __shared__ float A[512];
  for (int i = threadIdx.x; i < 512; i += 256) A[i] = attnW[(size_t)b * 512 + i];
  __syncthreads();
  if (pix >= N_PIX) return;
  const float* vb = v1 + (size_t)b * 64 * N_PIX;
  float o[64];
#pragma unroll
  for (int h = 0; h < 8; ++h) {
    float vv[8];
#pragma unroll
    for (int d = 0; d < 8; ++d) vv[d] = vb[(size_t)(h * 8 + d) * N_PIX + pix];
#pragma unroll
    for (int c = 0; c < 8; ++c) {
      float s = 0.f;
#pragma unroll
      for (int d = 0; d < 8; ++d) s = fmaf(A[h * 64 + c * 8 + d], vv[d], s);
      o[h * 8 + c] = s;
    }
  }
  float mu = 0.f;
#pragma unroll
  for (int ch = 0; ch < 64; ++ch) mu += o[ch];
  mu *= (1.f / 64.f);
  float var = 0.f;
#pragma unroll
  for (int ch = 0; ch < 64; ++ch) { const float t = o[ch] - mu; var = fmaf(t, t, var); }
  var *= (1.f / 64.f);
  const float inv = rsqrtf(var + 1e-5f);
  float* cb = cur + (size_t)b * 64 * N_PIX + pix;
#pragma unroll
  for (int ch = 0; ch < 64; ++ch)
    cb[(size_t)ch * N_PIX] += (o[ch] - mu) * inv * lnw[ch] + lnb[ch];
}

// ---------------------------------------------------------------------------
// final: out = cur + LayerNorm_c(m2)  (per pixel)
// ---------------------------------------------------------------------------
__global__ __launch_bounds__(256) void ln_residual_kernel(
    const float* __restrict__ m2, int m2_bstride, const float* __restrict__ cur,
    const float* __restrict__ lnw, const float* __restrict__ lnb,
    float* __restrict__ out) {
  const int b = blockIdx.y;
  const int pix = blockIdx.x * 256 + threadIdx.x;
  if (pix >= N_PIX) return;
  const float* mb = m2 + (size_t)b * m2_bstride + pix;
  float o[64];
#pragma unroll
  for (int ch = 0; ch < 64; ++ch) o[ch] = mb[(size_t)ch * N_PIX];
  float mu = 0.f;
#pragma unroll
  for (int ch = 0; ch < 64; ++ch) mu += o[ch];
  mu *= (1.f / 64.f);
  float var = 0.f;
#pragma unroll
  for (int ch = 0; ch < 64; ++ch) { const float t = o[ch] - mu; var = fmaf(t, t, var); }
  var *= (1.f / 64.f);
  const float inv = rsqrtf(var + 1e-5f);
  const float* cb = cur + (size_t)b * 64 * N_PIX + pix;
  float* ob = out + (size_t)b * 64 * N_PIX + pix;
#pragma unroll
  for (int ch = 0; ch < 64; ++ch)
    ob[(size_t)ch * N_PIX] = cb[(size_t)ch * N_PIX] + (o[ch] - mu) * inv * lnw[ch] + lnb[ch];
}

// ---------------------------------------------------------------------------
extern "C" void kernel_launch(void* const* d_in, const int* in_sizes, int n_in,
                              void* d_out, int out_size, void* d_ws, size_t ws_size,
                              hipStream_t stream) {
  const float* x      = (const float*)d_in[0];
  const float* a_qkv  = (const float*)d_in[1];
  const float* a_ce   = (const float*)d_in[2];
  const float* a_gd   = (const float*)d_in[3];
  const float* a_gd2  = (const float*)d_in[4];
  const float* a_ci   = (const float*)d_in[5];
  const float* a_hw   = (const float*)d_in[6];
  const float* b_qkv  = (const float*)d_in[7];
  const float* b_ce   = (const float*)d_in[8];
  const float* b_gd   = (const float*)d_in[9];
  const float* b_gd2  = (const float*)d_in[10];
  const float* b_ci   = (const float*)d_in[11];
  const float* b_hw   = (const float*)d_in[12];
  const float* fc1_w  = (const float*)d_in[13];
  const float* fc1_ce = (const float*)d_in[14];
  const float* fc1_gd = (const float*)d_in[15];
  const float* fc1_gd2= (const float*)d_in[16];
  const float* fc1_ci = (const float*)d_in[17];
  const float* fc2_w  = (const float*)d_in[18];
  const float* fc2_ce = (const float*)d_in[19];
  const float* fc2_gd = (const float*)d_in[20];
  const float* fc2_gd2= (const float*)d_in[21];
  const float* fc2_ci = (const float*)d_in[22];
  const float* ln1w = (const float*)d_in[23];
  const float* ln1b = (const float*)d_in[24];
  const float* ln2w = (const float*)d_in[25];
  const float* ln2b = (const float*)d_in[26];
  const float* ln3w = (const float*)d_in[27];
  const float* ln3b = (const float*)d_in[28];

  const int N = N_PIX;
  const size_t CN = (size_t)64 * N;
  float* ws   = (float*)d_ws;
  float* cur  = ws;                              // (8,64,N)
  float* qkv  = cur + 8 * CN;                    // (8,192,N): qkv / m / m2
  float* k1   = qkv + (size_t)8 * 192 * N;       // (8,64,N)
  float* v1   = k1 + 8 * CN;                     // (8,64,N)
  float* gmax = v1 + 8 * CN;                     // 512
  float* gl9  = gmax + 512;                      // 4608
  float* attnW = gl9 + 4608;                     // 4096
  _Float16* w16 = (_Float16*)(attnW + 4096);     // up to 8*64*576 halves

  hipMemcpyAsync(cur, x, 8 * CN * sizeof(float), hipMemcpyDeviceToDevice, stream);

  dim3 blk(256);
  dim3 gGemm192((N + 255) / 256, 12, 8);
  dim3 gGemm64((N + 255) / 256, 4, 8);
  dim3 gPix((N + 255) / 256, 8);
  const int flatKV = (int)((8 * CN + 255) / 256);

  // ---- attention block A -------------------------------------------------
  channel_max_kernel<<<512, blk, 0, stream>>>(cur, gmax, 64 * N);
  gates1x1_kernel<<<8, blk, 0, stream>>>(gmax, a_ce, a_gd, a_gd2, a_ci, a_qkv, w16, 192, 24);
  gemm_wmma_kernel<192, 64, false, 0><<<gGemm192, blk, 0, stream>>>(w16, cur, qkv, 64 * N, 192 * N);
  kv_accum_kernel<<<flatKV, blk, 0, stream>>>(qkv, k1, v1, 0);
  attn_scores_kernel<<<64, blk, 0, stream>>>(qkv, k1, a_hw, attnW);
  attn_apply_ln_kernel<<<gPix, blk, 0, stream>>>(attnW, v1, ln1w, ln1b, cur);

  // ---- attention block B (k1+=, v1+=) ------------------------------------
  channel_max_kernel<<<512, blk, 0, stream>>>(cur, gmax, 64 * N);
  gates1x1_kernel<<<8, blk, 0, stream>>>(gmax, b_ce, b_gd, b_gd2, b_ci, b_qkv, w16, 192, 24);
  gemm_wmma_kernel<192, 64, false, 0><<<gGemm192, blk, 0, stream>>>(w16, cur, qkv, 64 * N, 192 * N);
  kv_accum_kernel<<<flatKV, blk, 0, stream>>>(qkv, k1, v1, 1);
  attn_scores_kernel<<<64, blk, 0, stream>>>(qkv, k1, b_hw, attnW);
  attn_apply_ln_kernel<<<gPix, blk, 0, stream>>>(attnW, v1, ln2w, ln2b, cur);

  // ---- FFN ---------------------------------------------------------------
  channel_max_kernel<<<512, blk, 0, stream>>>(cur, gmax, 64 * N);
  gates1x1_kernel<<<8, blk, 0, stream>>>(gmax, fc1_ce, fc1_gd, fc1_gd2, fc1_ci, fc1_w, w16, 64, 8);
  // m = GELU(fc1(cur)) stored in qkv rows 0..63 (batch stride 192*N)
  gemm_wmma_kernel<64, 64, false, 1><<<gGemm64, blk, 0, stream>>>(w16, cur, qkv, 64 * N, 192 * N);
  pool3_kernel<<<512, blk, 0, stream>>>(qkv, gl9, 192 * N);
  gates3x3_kernel<<<8, blk, 0, stream>>>(gl9, fc2_ce, fc2_gd, fc2_gd2, fc2_ci, fc2_w, w16);
  // m2 = fc2(m) via im2col GEMM, stored in qkv rows 64..127
  gemm_wmma_kernel<64, 576, true, 0><<<gGemm64, blk, 0, stream>>>(w16, qkv, qkv + 64 * N, 192 * N, 192 * N);
  ln_residual_kernel<<<gPix, blk, 0, stream>>>(qkv + 64 * N, 192 * N, cur, ln3w, ln3b, (float*)d_out);

  (void)in_sizes; (void)n_in; (void)out_size; (void)ws_size;
}
